// MaskedScaledDotProduct_55087250538575
// MI455X (gfx1250) — compile-verified, matrix operand order checked by reading, not run on hardware
//
#include <hip/hip_runtime.h>

typedef _Float16 v16h __attribute__((ext_vector_type(16)));
typedef _Float16 v8h  __attribute__((ext_vector_type(8)));
typedef _Float16 v4h  __attribute__((ext_vector_type(4)));
typedef float    v8f  __attribute__((ext_vector_type(8)));
typedef float    v4f  __attribute__((ext_vector_type(4)));

#define B_ 128
#define T_ 256
#define C_ 512
#define H_ 512
#define WPB 2   // waves per block in attention kernel

__device__ inline v16h make_frag(v8h lo, v8h hi) {
  v16h r;
#pragma unroll
  for (int i = 0; i < 8; ++i) { r[i] = lo[i]; r[i + 8] = hi[i]; }
  return r;
}

__device__ inline v16h load_frag(const _Float16* p) {
  return make_frag(*(const v8h*)p, *(const v8h*)(p + 16));
}

// ---------------------------------------------------------------------------
// Kernel 0: one-pass fp32 -> f16 conversion of x and the three weights.
// ---------------------------------------------------------------------------
__global__ __launch_bounds__(256) void cvt_f32_to_f16(
    const float* __restrict__ x,
    const float* __restrict__ Wq,
    const float* __restrict__ Wk,
    const float* __restrict__ Wv,
    _Float16* __restrict__ xh,
    _Float16* __restrict__ Wh)
{
  const size_t Nx4 = (size_t)B_ * T_ * C_ / 4;     // 4,194,304 float4 groups
  const size_t Nw4 = (size_t)H_ * C_ / 4;          // 65,536 per weight
  size_t t = (size_t)blockIdx.x * 256 + threadIdx.x;

  const float* src; _Float16* dst; size_t off;
  if (t < Nx4) {
    src = x; dst = xh; off = t;
  } else {
    size_t r = t - Nx4;
    if (r >= 3 * Nw4) return;
    int ws = (int)(r / Nw4); off = r % Nw4;
    src = (ws == 0) ? Wq : (ws == 1) ? Wk : Wv;
    dst = Wh + (size_t)ws * H_ * C_;
  }
  v4f v = *(const v4f*)(src + off * 4);
  v4h h;
#pragma unroll
  for (int i = 0; i < 4; ++i) h[i] = (_Float16)v[i];
  *(v4h*)(dst + off * 4) = h;
}

// ---------------------------------------------------------------------------
// Kernel 1: Q/K/V = xh @ Wh^T (pure f16 WMMA). One wave computes a 16x64
// strip: one A fragment feeds 4 WMMAs. Software-pipelined: fragments for
// step kc+1 are loaded before the WMMAs of step kc execute.
// V stored transposed as Vt[b][h][t] (contiguous b128 store per lane).
// ---------------------------------------------------------------------------
__global__ __launch_bounds__(256) void qkv_gemm_f16(
    const _Float16* __restrict__ xh,
    const _Float16* __restrict__ Wh,
    _Float16* __restrict__ Qh,
    _Float16* __restrict__ Kh,
    _Float16* __restrict__ Vt)
{
  const int lane = threadIdx.x & 31;
  const int wave = threadIdx.x >> 5;
  const int gw   = blockIdx.x * 8 + wave;
  const int tiles_m = (B_ * T_) / 16;              // 2048
  const int ngroups = H_ / 64;                     // 8 groups of 4 N-tiles
  const int w   = gw / (tiles_m * ngroups);        // 0=Q 1=K 2=V
  const int rem = gw % (tiles_m * ngroups);
  const int mt  = rem / ngroups;
  const int ng  = rem % ngroups;
  const int m0  = mt * 16, n0 = ng * 64;

  const int row  = lane & 15;
  const int koff = (lane >= 16) ? 8 : 0;

  const _Float16* arow = xh + (size_t)(m0 + row) * C_ + koff;
  const _Float16* brow = Wh + (size_t)w * H_ * C_ + (size_t)(n0 + row) * C_ + koff;

  v8f acc[4] = {};
  v16h af = load_frag(arow);
  v16h bf[4];
#pragma unroll
  for (int t = 0; t < 4; ++t) bf[t] = load_frag(brow + (size_t)t * 16 * C_);

#pragma unroll
  for (int kc = 0; kc < C_ / 32; ++kc) {
    v16h afn; v16h bfn[4];
    if (kc + 1 < C_ / 32) {                        // prefetch next step's frags
      afn = load_frag(arow + (kc + 1) * 32);
#pragma unroll
      for (int t = 0; t < 4; ++t)
        bfn[t] = load_frag(brow + (size_t)t * 16 * C_ + (kc + 1) * 32);
    }
#pragma unroll
    for (int t = 0; t < 4; ++t)
      acc[t] = __builtin_amdgcn_wmma_f32_16x16x32_f16(false, af, false, bf[t],
                                                      (short)0, acc[t], false, false);
    if (kc + 1 < C_ / 32) {
      af = afn;
#pragma unroll
      for (int t = 0; t < 4; ++t) bf[t] = bfn[t];
    }
  }

  const int col    = lane & 15;
  const int rowoff = (lane >= 16) ? 8 : 0;
#pragma unroll
  for (int t = 0; t < 4; ++t) {
    const int nt0 = n0 + t * 16;
    if (w < 2) {
      _Float16* O = (w == 0) ? Qh : Kh;
#pragma unroll
      for (int i = 0; i < 8; ++i)
        O[(size_t)(m0 + rowoff + i) * H_ + nt0 + col] = (_Float16)acc[t][i];
    } else {
      const int b  = m0 / T_;
      const int t0 = m0 % T_;
      v8h pk;
#pragma unroll
      for (int i = 0; i < 8; ++i) pk[i] = (_Float16)acc[t][i];
      *(v8h*)(Vt + (size_t)b * H_ * T_ + (size_t)(nt0 + col) * T_ + t0 + rowoff) = pk;
    }
  }
}

// ---------------------------------------------------------------------------
// Kernel 2: causal attention. One wave per (batch, 16-query tile).
// Q tile held in 128 VGPRs; S via WMMA -> LDS; 2-lane/row softmax with
// __shfl_xor combines; P fragments cached in registers across all 32 N-tiles.
// K and V fragment loads are software-pipelined against the WMMAs.
// ---------------------------------------------------------------------------
__global__ __launch_bounds__(32 * WPB) void attn_f16(
    const _Float16* __restrict__ Qh,
    const _Float16* __restrict__ Kh,
    const _Float16* __restrict__ Vt,
    float* __restrict__ out)
{
  __shared__ float    Sbuf[WPB][16 * 256];
  __shared__ _Float16 Pbuf[WPB][16 * 256];

  const int lane = threadIdx.x & 31;
  const int wave = threadIdx.x >> 5;
  const int gw = blockIdx.x * WPB + wave;          // 0..2047
  const int b  = gw >> 4;
  const int qi = gw & 15;
  const int q0 = qi * 16;

  float*    S = &Sbuf[wave][0];
  _Float16* P = &Pbuf[wave][0];

  const int row    = lane & 15;
  const int koff   = (lane >= 16) ? 8 : 0;
  const int rowoff = (lane >= 16) ? 8 : 0;
  const float scale = 0.044194173824159216f;       // 1/sqrt(512)

  // ---- load the whole Q tile into registers (16 chunks x 8 VGPRs) ----
  const _Float16* qrow = Qh + ((size_t)b * T_ + q0 + row) * H_ + koff;
  v16h qf[16];
#pragma unroll
  for (int c = 0; c < 16; ++c) qf[c] = load_frag(qrow + c * 32);

  // ---- scores: key tiles j <= qi (causal), pipelined K loads ----
  const _Float16* kbase = Kh + ((size_t)b * T_ + row) * H_ + koff;
  for (int j = 0; j <= qi; ++j) {
    const _Float16* krow = kbase + (size_t)j * 16 * H_;
    if (j < qi) __builtin_prefetch(krow + (size_t)16 * H_, 0, 0);
    v8f acc = {};
    v16h bf = load_frag(krow);
#pragma unroll
    for (int c = 0; c < H_ / 32; ++c) {
      v16h bfn;
      if (c + 1 < H_ / 32) bfn = load_frag(krow + (c + 1) * 32);
      acc = __builtin_amdgcn_wmma_f32_16x16x32_f16(false, qf[c], false, bf,
                                                   (short)0, acc, false, false);
      if (c + 1 < H_ / 32) bf = bfn;
    }
#pragma unroll
    for (int i = 0; i < 8; ++i) {
      float v = acc[i] * scale;
      const int m = i + rowoff;
      const int n = row;
      if (j == qi && n > m) v = -INFINITY;
      S[m * 256 + j * 16 + n] = v;
    }
  }

  // ---- softmax: lanes (r, r+16) split each row's columns in half ----
  const int nvalid = (qi + 1) * 16;
  const int kpad   = (nvalid + 31) & ~31;
  {
    const int half = lane >> 4;
    const int nv2  = nvalid >> 1;
    const int c0   = half * nv2;
    float* Sr = S + row * 256;
    float mx = -INFINITY;
    for (int c = 0; c < nv2; ++c) mx = fmaxf(mx, Sr[c0 + c]);
    mx = fmaxf(mx, __shfl_xor(mx, 16, 32));
    float sum = 0.f;
    for (int c = 0; c < nv2; ++c) {
      float e = __expf(Sr[c0 + c] - mx); Sr[c0 + c] = e; sum += e;
    }
    sum += __shfl_xor(sum, 16, 32);
    const float inv = 1.0f / sum;
    _Float16* Pr = P + row * 256;
    for (int c = 0; c < nv2; ++c) Pr[c0 + c] = (_Float16)(Sr[c0 + c] * inv);
    if (half == 1)                                   // zero-pad to 32-key multiple
      for (int c = nvalid; c < kpad; ++c) Pr[c] = (_Float16)0.f;
  }

  // ---- cache P fragments in registers, then O = P @ V (pipelined V loads) --
  const int kchunks = kpad / 32;                    // wave-uniform, 1..8
  const _Float16* prow = P + row * 256 + koff;
  v16h pf[8];
#pragma unroll
  for (int kc = 0; kc < 8; ++kc)
    if (kc < kchunks) pf[kc] = load_frag(prow + kc * 32);

  const int col = row;
  for (int nt = 0; nt < H_ / 16; ++nt) {
    const _Float16* vcol =
        Vt + (size_t)b * H_ * T_ + (size_t)(nt * 16 + col) * T_ + koff;
    v8f acc = {};
    v16h bf = load_frag(vcol);                      // kchunks >= 1 always
#pragma unroll
    for (int kc = 0; kc < 8; ++kc) {
      if (kc < kchunks) {
        v16h bfn;
        if (kc + 1 < kchunks) bfn = load_frag(vcol + (kc + 1) * 32);
        acc = __builtin_amdgcn_wmma_f32_16x16x32_f16(false, pf[kc], false, bf,
                                                     (short)0, acc, false, false);
        if (kc + 1 < kchunks) bf = bfn;
      }
    }
#pragma unroll
    for (int i = 0; i < 8; ++i)
      out[((size_t)b * T_ + q0 + rowoff + i) * H_ + nt * 16 + col] = acc[i];
  }
}

// ---------------------------------------------------------------------------
extern "C" void kernel_launch(void* const* d_in, const int* in_sizes, int n_in,
                              void* d_out, int out_size, void* d_ws, size_t ws_size,
                              hipStream_t stream) {
  (void)in_sizes; (void)n_in; (void)out_size; (void)ws_size;

  const float* x  = (const float*)d_in[0];
  const float* Wq = (const float*)d_in[1];
  const float* Wk = (const float*)d_in[2];
  const float* Wv = (const float*)d_in[3];
  float* out = (float*)d_out;

  const size_t nqkv = (size_t)B_ * T_ * H_;        // 16,777,216 elems
  _Float16* Qh = (_Float16*)d_ws;
  _Float16* Kh = Qh + nqkv;
  _Float16* Vt = Kh + nqkv;
  _Float16* xh = Vt + nqkv;
  _Float16* Wh = xh + (size_t)B_ * T_ * C_;        // 3*H*C halves; ~130 MB total

  // Kernel 0: convert x and weights to f16
  const size_t Ncvt = (size_t)B_ * T_ * C_ / 4 + 3 * (size_t)H_ * C_ / 4;
  dim3 blk0(256), grid0((unsigned)((Ncvt + 255) / 256));
  cvt_f32_to_f16<<<grid0, blk0, 0, stream>>>(x, Wq, Wk, Wv, xh, Wh);

  // Kernel 1: 2048 M-tiles * 8 N-groups * 3 weights = 49152 waves / 8 per block
  dim3 blk1(256), grid1((B_ * T_ / 16) * (H_ / 64) * 3 / 8);
  qkv_gemm_f16<<<grid1, blk1, 0, stream>>>(xh, Wh, Qh, Kh, Vt);

  // Kernel 2: 2048 query-tile waves / WPB per block
  dim3 blk2(32 * WPB), grid2(B_ * (T_ / 16) / WPB);
  attn_f16<<<grid2, blk2, 0, stream>>>(Qh, Kh, Vt, out);
}
